// Dual_UFormer_Image_Deraining_Network_31112743092356
// MI455X (gfx1250) — matrix operand order repaired and checked by reading, not compiled, not run on hardware
//
#include <hip/hip_runtime.h>
#include <hip/hip_bf16.h>
#include <math.h>

typedef __attribute__((ext_vector_type(16))) _Float16 v16h;
typedef __attribute__((ext_vector_type(8)))  float    v8f;

#define BN 2          // batch
#define CCH 48        // channels
#define HEADS 8
#define HDIM 6        // 48/8
#define HID 127       // int(48*2.66)

__device__ __constant__ float cU1P1[16] = {-0.5f,-0.5f,-0.5f,0.5f,  0.5f,-0.5f,0.5f,0.5f,
                                            0.5f,0.5f,-0.5f,0.5f,  0.5f,-0.5f,-0.5f,-0.5f};
__device__ __constant__ float cU2P2[16] = {-1.f,1.f,1.f,-1.f,  -1.f,-1.f,-1.f,-1.f,
                                           -1.f,-1.f,1.f,1.f,   1.f,-1.f,1.f,-1.f};
__device__ __constant__ float cP2U2[16] = {-1.f,-1.f,-1.f,1.f,  1.f,-1.f,-1.f,-1.f,
                                            1.f,-1.f,1.f,1.f,  -1.f,-1.f,1.f,-1.f};
__device__ __constant__ float cP1U1[16] = {-0.125f,0.125f,0.125f,0.125f,  -0.125f,-0.125f,0.125f,-0.125f,
                                           -0.125f,0.125f,-0.125f,-0.125f, 0.125f,0.125f,0.125f,-0.125f};

// ---------------- WMMA GEMM: Out[b,M,n] = W[M,K] * X[b,K,n] (+ Res) ----------------
// grid: (n/256, ceil(M/16), B), block 256 (8 waves). Wave w -> 32 cols at blockIdx.x*256+w*32.
// A tile (16 rows of W, K padded to 128) staged once in LDS as f16, ZERO for k>=K.
// B rows for k>=K are loaded unguarded (multiplied by zero A entries); X buffers have
// live workspace behind them so the over-read is within the allocation and finite.
__global__ void k_gemm_wmma(const float* __restrict__ Wt, const float* __restrict__ X,
                            const float* __restrict__ Res, float* __restrict__ Out,
                            int M, int K, int n) {
  __shared__ _Float16 As[16 * 128];
  const int tid  = threadIdx.x;
  const int wave = tid >> 5;
  const int lane = tid & 31;
  const int m0 = blockIdx.y * 16;
  const int n0 = blockIdx.x * 256 + wave * 32;
  const size_t nn = (size_t)n;
  const size_t bX = (size_t)blockIdx.z * (size_t)K * nn;
  const size_t bO = (size_t)blockIdx.z * (size_t)M * nn;

  for (int i = tid; i < 16 * 128; i += 256) {
    int m = i >> 7, k = i & 127;
    float v = 0.0f;
    if (k < K && (m0 + m) < M) v = Wt[(size_t)(m0 + m) * K + k];
    As[i] = (_Float16)v;
  }
  __syncthreads();

  const int ml = lane & 15;    // A row in tile / D column selector
  const int kh = lane >> 4;    // K-half selector
  const uint32_t* As32 = (const uint32_t*)As;
  // per-lane base pointers: column n0+ml (tile 0) and n0+ml+16 (tile 1), K-half folded in
  const float* xp0 = X + bX + (size_t)(n0 + ml) + (size_t)(kh * 16) * nn;
  const float* xp1 = xp0 + 16;

  v8f acc0 = {}, acc1 = {};
  const int KT = (K + 31) >> 5;

  for (int kk = 0; kk < KT; ++kk) {
    const float* r0 = xp0 + (size_t)(kk * 32) * nn;
    const float* r1 = xp1 + (size_t)(kk * 32) * nn;
    union { v16h h; uint32_t u[8]; } a;
    float f0[16], f1[16];
#pragma unroll
    for (int j = 0; j < 8; ++j) {
      f0[2 * j]     = r0[(size_t)(2 * j) * nn];
      f0[2 * j + 1] = r0[(size_t)(2 * j + 1) * nn];
      f1[2 * j]     = r1[(size_t)(2 * j) * nn];
      f1[2 * j + 1] = r1[(size_t)(2 * j + 1) * nn];
      // A in this kernel's contiguous k order (same as B): VGPR j holds K pair (16*kh + 2*j)
      int kpA = kk * 32 + (kh << 4) + (j << 1);
      a.u[j] = As32[(ml * 128 + kpA) >> 1];
    }
    __builtin_prefetch(r0 + (size_t)32 * nn, 0, 1);   // speculative next K-slab
    v16h b0, b1;
#pragma unroll
    for (int e = 0; e < 16; ++e) { b0[e] = (_Float16)f0[e]; b1[e] = (_Float16)f1[e]; }
    acc0 = __builtin_amdgcn_wmma_f32_16x16x32_f16(false, a.h, false, b0, (short)0, acc0, false, false);
    acc1 = __builtin_amdgcn_wmma_f32_16x16x32_f16(false, a.h, false, b1, (short)0, acc1, false, false);
  }

#pragma unroll
  for (int r = 0; r < 8; ++r) {
    int m = m0 + r + (kh << 3);
    if (m < M) {
      size_t o = bO + (size_t)m * nn + (size_t)(n0 + ml);
      float v0 = acc0[r], v1 = acc1[r];
      if (Res) { v0 += Res[o]; v1 += Res[o + 16]; }
      Out[o] = v0;
      Out[o + 16] = v1;
    }
  }
}

// ---------------- LayerNorm over channel dim (C=48) ----------------
__global__ void k_layernorm(const float* __restrict__ x, const float* __restrict__ w,
                            const float* __restrict__ bi, float* __restrict__ out, int n) {
  int p = blockIdx.x * 256 + threadIdx.x;
  size_t base = (size_t)blockIdx.y * CCH * (size_t)n + p;
  float s = 0.f, s2 = 0.f;
#pragma unroll
  for (int c = 0; c < CCH; ++c) { float v = x[base + (size_t)c * n]; s += v; s2 += v * v; }
  float mu  = s * (1.0f / CCH);
  float var = s2 * (1.0f / CCH) - mu * mu;
  float inv = rsqrtf(var + 1e-5f);
#pragma unroll
  for (int c = 0; c < CCH; ++c) {
    float v = x[base + (size_t)c * n];
    out[base + (size_t)c * n] = (v - mu) * inv * w[c] + bi[c];
  }
}

// ---------------- depthwise 3x3, pad 1 ----------------
__global__ void k_dwconv3(const float* __restrict__ in, const float* __restrict__ w,
                          float* __restrict__ out, int Cn, int Hh, int Ww) {
  int p = blockIdx.x * 256 + threadIdx.x;
  int c = blockIdx.y, b = blockIdx.z;
  int y = p / Ww, x = p % Ww;
  size_t base = ((size_t)b * Cn + c) * (size_t)(Hh * Ww);
  const float* wc = w + (size_t)c * 9;
  float acc = 0.f;
#pragma unroll
  for (int ky = 0; ky < 3; ++ky) {
    int yy = y + ky - 1; if (yy < 0 || yy >= Hh) continue;
#pragma unroll
    for (int kx = 0; kx < 3; ++kx) {
      int xx = x + kx - 1; if (xx < 0 || xx >= Ww) continue;
      acc += wc[ky * 3 + kx] * in[base + (size_t)yy * Ww + xx];
    }
  }
  out[base + p] = acc;
}

// ---------------- q/k row L2 norms: rows 0..47=q, 48..95=k ----------------
__global__ void k_rownorm(const float* __restrict__ qkv, float* __restrict__ nrm, int n) {
  __shared__ float sd[256];
  int blk = blockIdx.x, b = blk / 96, row = blk % 96;
  const float* d = qkv + ((size_t)b * 144 + row) * (size_t)n;
  float s = 0.f;
  for (int i = threadIdx.x; i < n; i += 256) { float v = d[i]; s += v * v; }
  sd[threadIdx.x] = s; __syncthreads();
  for (int st = 128; st > 0; st >>= 1) { if (threadIdx.x < st) sd[threadIdx.x] += sd[threadIdx.x + st]; __syncthreads(); }
  if (threadIdx.x == 0) nrm[blk] = sqrtf(sd[0]);
}

// ---------------- raw q.k dot per (b,h,c,d) ----------------
__global__ void k_qkdot(const float* __restrict__ qkv, float* __restrict__ sdot, int n) {
  __shared__ float sd[256];
  int cd = blockIdx.x, h = blockIdx.y, b = blockIdx.z;
  int c = cd / HDIM, dd = cd % HDIM;
  const float* q = qkv + ((size_t)b * 144 + h * HDIM + c) * (size_t)n;
  const float* k = qkv + ((size_t)b * 144 + CCH + h * HDIM + dd) * (size_t)n;
  float s = 0.f;
  for (int i = threadIdx.x; i < n; i += 256) s += q[i] * k[i];
  sd[threadIdx.x] = s; __syncthreads();
  for (int st = 128; st > 0; st >>= 1) { if (threadIdx.x < st) sd[threadIdx.x] += sd[threadIdx.x + st]; __syncthreads(); }
  if (threadIdx.x == 0) sdot[(((size_t)b * HEADS + h) * HDIM + c) * HDIM + dd] = sd[0];
}

// ---------------- softmax over 6x6 with norm + temperature ----------------
__global__ void k_softmax(const float* __restrict__ sdot, const float* __restrict__ nrm,
                          const float* __restrict__ temp, float* __restrict__ attn) {
  int h = blockIdx.x, b = blockIdx.y, c = threadIdx.x;
  if (c >= HDIM) return;
  float nq = fmaxf(nrm[b * 96 + h * HDIM + c], 1e-12f);
  float t = temp[h];
  float lg[HDIM]; float mx = -3.4e38f;
#pragma unroll
  for (int d = 0; d < HDIM; ++d) {
    float nk = fmaxf(nrm[b * 96 + CCH + h * HDIM + d], 1e-12f);
    float v = sdot[(((size_t)b * HEADS + h) * HDIM + c) * HDIM + d] / (nq * nk) * t;
    lg[d] = v; mx = fmaxf(mx, v);
  }
  float se = 0.f;
#pragma unroll
  for (int d = 0; d < HDIM; ++d) { lg[d] = expf(lg[d] - mx); se += lg[d]; }
#pragma unroll
  for (int d = 0; d < HDIM; ++d)
    attn[(((size_t)b * HEADS + h) * HDIM + c) * HDIM + d] = lg[d] / se;
}

// ---------------- out = attn @ v ----------------
__global__ void k_attnv(const float* __restrict__ qkv, const float* __restrict__ attn,
                        float* __restrict__ out, int n) {
  int p = blockIdx.x * 256 + threadIdx.x;
  int co = blockIdx.y, b = blockIdx.z;
  int h = co / HDIM, c = co % HDIM;
  const float* a = attn + (((size_t)b * HEADS + h) * HDIM + c) * HDIM;
  const float* v = qkv + ((size_t)b * 144 + 2 * CCH + h * HDIM) * (size_t)n;
  float acc = 0.f;
#pragma unroll
  for (int d = 0; d < HDIM; ++d) acc += a[d] * v[(size_t)d * n + p];
  out[((size_t)b * CCH + co) * (size_t)n + p] = acc;
}

// ---------------- GEGLU gate: g = gelu(x1)*x2, hidden=127 ----------------
__global__ void k_geglu(const float* __restrict__ pd, float* __restrict__ g, int n) {
  int p = blockIdx.x * 256 + threadIdx.x;
  int c = blockIdx.y, b = blockIdx.z;
  const float* base = pd + (size_t)b * (2 * HID) * (size_t)n;
  float x1v = base[(size_t)c * n + p];
  float x2v = base[(size_t)(HID + c) * n + p];
  float ge = 0.5f * x1v * (1.0f + erff(x1v * 0.7071067811865475f));
  g[((size_t)b * HID + c) * (size_t)n + p] = ge * x2v;
}

// ---------------- NLWT: x(B,48,2h1,2w1) -> A,B,C,D (B,48,h1,w1) ----------------
__global__ void k_nlwt(const float* __restrict__ x, float* __restrict__ A, float* __restrict__ Bo,
                       float* __restrict__ Co, float* __restrict__ Do, int h1, int w1) {
  int p = blockIdx.x * 256 + threadIdx.x;
  int c = blockIdx.y, b = blockIdx.z;
  int i = p / w1, j = p % w1;
  int W2 = 2 * w1;
  size_t xb = ((size_t)b * CCH + c) * (size_t)(4 * h1 * w1);
  auto x1f = [&](int k, int ii, int jj) -> float {
    float c00 = x[xb + (size_t)(2 * ii) * W2 + 2 * jj];
    float c01 = x[xb + (size_t)(2 * ii) * W2 + 2 * jj + 1];
    float c10 = x[xb + (size_t)(2 * ii + 1) * W2 + 2 * jj];
    float c11 = x[xb + (size_t)(2 * ii + 1) * W2 + 2 * jj + 1];
    return cU1P1[k * 4 + 0] * c00 + cU1P1[k * 4 + 1] * c01 +
           cU1P1[k * 4 + 2] * c10 + cU1P1[k * 4 + 3] * c11;
  };
  int ip = (i + 1 == h1) ? 0 : i + 1;
  int jp = (j + 1 == w1) ? 0 : j + 1;
  float t0 = x1f(0, i, j), t1 = x1f(1, ip, j), t2 = x1f(2, i, jp), t3 = x1f(3, ip, jp);
  size_t ob = ((size_t)b * CCH + c) * (size_t)(h1 * w1) + p;
  A[ob]  = cU2P2[0]  * t0 + cU2P2[1]  * t1 + cU2P2[2]  * t2 + cU2P2[3]  * t3;
  Bo[ob] = cU2P2[4]  * t0 + cU2P2[5]  * t1 + cU2P2[6]  * t2 + cU2P2[7]  * t3;
  Co[ob] = cU2P2[8]  * t0 + cU2P2[9]  * t1 + cU2P2[10] * t2 + cU2P2[11] * t3;
  Do[ob] = cU2P2[12] * t0 + cU2P2[13] * t1 + cU2P2[14] * t2 + cU2P2[15] * t3;
}

// ---------------- INLWT: A,B,C,D (B,48,h1,w1) -> out (B,48,2h1,2w1) ----------------
__global__ void k_inlwt(const float* __restrict__ A, const float* __restrict__ Bo,
                        const float* __restrict__ Co, const float* __restrict__ Do,
                        float* __restrict__ out, int h1, int w1) {
  int p = blockIdx.x * 256 + threadIdx.x;
  int c = blockIdx.y, b = blockIdx.z;
  int i = p / w1, j = p % w1;
  size_t ib = ((size_t)b * CCH + c) * (size_t)(h1 * w1);
  auto Y2f = [&](int m, int ii, int jj) -> float {
    size_t o = ib + (size_t)ii * w1 + jj;
    return cP2U2[m * 4 + 0] * A[o] + cP2U2[m * 4 + 1] * Bo[o] +
           cP2U2[m * 4 + 2] * Co[o] + cP2U2[m * 4 + 3] * Do[o];
  };
  int im = (i == 0) ? h1 - 1 : i - 1;
  int jm = (j == 0) ? w1 - 1 : j - 1;
  float t0 = Y2f(0, i, j), t1 = Y2f(1, im, j), t2 = Y2f(2, i, jm), t3 = Y2f(3, im, jm);
  int W2 = 2 * w1;
  size_t ob = ((size_t)b * CCH + c) * (size_t)(4 * h1 * w1);
#pragma unroll
  for (int pp = 0; pp < 2; ++pp)
#pragma unroll
    for (int qq = 0; qq < 2; ++qq) {
      int k = pp * 2 + qq;
      float v = cP1U1[k * 4 + 0] * t0 + cP1U1[k * 4 + 1] * t1 +
                cP1U1[k * 4 + 2] * t2 + cP1U1[k * 4 + 3] * t3;
      out[ob + (size_t)(2 * i + pp) * W2 + (2 * j + qq)] = v;
    }
}

// ---------------- GFM: per-(b,ch of 96) mean & max over pixels ----------------
__global__ void k_chanreduce(const float* __restrict__ b1, const float* __restrict__ b2,
                             float* __restrict__ avg, float* __restrict__ mx, int n) {
  __shared__ float ss[256], sm[256];
  int blk = blockIdx.x, b = blk / 96, ch = blk % 96;
  const float* d = (ch < CCH) ? (b1 + ((size_t)b * CCH + ch) * (size_t)n)
                              : (b2 + ((size_t)b * CCH + (ch - CCH)) * (size_t)n);
  float s = 0.f, m = -3.4e38f;
  for (int i = threadIdx.x; i < n; i += 256) { float v = d[i]; s += v; m = fmaxf(m, v); }
  ss[threadIdx.x] = s; sm[threadIdx.x] = m; __syncthreads();
  for (int st = 128; st > 0; st >>= 1) {
    if (threadIdx.x < st) {
      ss[threadIdx.x] += ss[threadIdx.x + st];
      sm[threadIdx.x] = fmaxf(sm[threadIdx.x], sm[threadIdx.x + st]);
    }
    __syncthreads();
  }
  if (threadIdx.x == 0) { avg[blk] = ss[0] / (float)n; mx[blk] = sm[0]; }
}

// ---------------- GFM MLP: cw = sigmoid(w2@relu(w1@avg) + w2@relu(w1@mx)) ----------------
__global__ void k_gfm_mlp(const float* __restrict__ avg, const float* __restrict__ mxv,
                          const float* __restrict__ w1, const float* __restrict__ w2,
                          float* __restrict__ cw) {
  __shared__ float ha[6], hm[6];
  int b = blockIdx.x, t = threadIdx.x;
  if (t < 6) {
    float s = 0.f;
    for (int i = 0; i < 96; ++i) s += w1[t * 96 + i] * avg[b * 96 + i];
    ha[t] = fmaxf(s, 0.f);
  } else if (t < 12) {
    int tt = t - 6; float s = 0.f;
    for (int i = 0; i < 96; ++i) s += w1[tt * 96 + i] * mxv[b * 96 + i];
    hm[tt] = fmaxf(s, 0.f);
  }
  __syncthreads();
  float s = 0.f;
#pragma unroll
  for (int k = 0; k < 6; ++k) s += w2[t * 6 + k] * (ha[k] + hm[k]);
  cw[b * 96 + t] = 1.0f / (1.0f + expf(-s));
}

// ---------------- GFM combine + per-pixel channel mean/max ----------------
__global__ void k_gfm_combine(const float* __restrict__ b1, const float* __restrict__ b2,
                              const float* __restrict__ cw, float* __restrict__ comb,
                              float* __restrict__ savg, float* __restrict__ smax, int n) {
  int p = blockIdx.x * 256 + threadIdx.x;
  int b = blockIdx.y;
  float s = 0.f, m = -3.4e38f;
#pragma unroll 4
  for (int c = 0; c < CCH; ++c) {
    size_t o = ((size_t)b * CCH + c) * (size_t)n + p;
    float v = cw[b * 96 + c] * b1[o] + cw[b * 96 + CCH + c] * b2[o];
    comb[o] = v; s += v; m = fmaxf(m, v);
  }
  savg[(size_t)b * n + p] = s * (1.0f / CCH);
  smax[(size_t)b * n + p] = m;
}

// ---------------- 7x7 spatial attention conv (2-ch in, 1-ch out), sigmoid ----------------
__global__ void k_saconv(const float* __restrict__ savg, const float* __restrict__ smax,
                         const float* __restrict__ w, const float* __restrict__ bias,
                         float* __restrict__ samap, int Hh, int Ww) {
  int p = blockIdx.x * 256 + threadIdx.x;
  int b = blockIdx.y;
  int y = p / Ww, x = p % Ww;
  size_t bn = (size_t)b * Hh * Ww;
  float acc = bias[0];
  for (int ky = 0; ky < 7; ++ky) {
    int yy = y + ky - 3; if (yy < 0 || yy >= Hh) continue;
    for (int kx = 0; kx < 7; ++kx) {
      int xx = x + kx - 3; if (xx < 0 || xx >= Ww) continue;
      size_t o = bn + (size_t)yy * Ww + xx;
      acc += w[ky * 7 + kx] * savg[o] + w[49 + ky * 7 + kx] * smax[o];
    }
  }
  samap[bn + p] = 1.0f / (1.0f + expf(-acc));
}

// ---------------- final: out = sigmoid(sa) * comb ----------------
__global__ void k_final(const float* __restrict__ comb, const float* __restrict__ samap,
                        float* __restrict__ out, int n) {
  int p = blockIdx.x * 256 + threadIdx.x;
  int c = blockIdx.y, b = blockIdx.z;
  size_t o = ((size_t)b * CCH + c) * (size_t)n + p;
  out[o] = comb[o] * samap[(size_t)b * n + p];
}

extern "C" void kernel_launch(void* const* d_in, const int* in_sizes, int n_in,
                              void* d_out, int out_size, void* d_ws, size_t ws_size,
                              hipStream_t stream) {
  (void)in_sizes; (void)n_in; (void)out_size; (void)ws_size;
  const float* x        = (const float*)d_in[0];
  const float* ln1_w    = (const float*)d_in[1];
  const float* ln1_b    = (const float*)d_in[2];
  const float* temp     = (const float*)d_in[3];
  const float* qkv_w    = (const float*)d_in[4];
  const float* qkv_dw_w = (const float*)d_in[5];
  const float* proj_w   = (const float*)d_in[6];
  const float* ln2_w    = (const float*)d_in[7];
  const float* ln2_b    = (const float*)d_in[8];
  const float* pin_w    = (const float*)d_in[9];
  const float* ffn_dw_w = (const float*)d_in[10];
  const float* pout_w   = (const float*)d_in[11];
  const float* ca_w1    = (const float*)d_in[12];
  const float* ca_w2    = (const float*)d_in[13];
  const float* sa_w     = (const float*)d_in[14];
  const float* sa_b     = (const float*)d_in[15];
  float* outp = (float*)d_out;

  const int n1 = 65536, n2 = 16384, h1 = 128;
  const size_t szC = (size_t)BN * CCH * n1;       // 6.29M floats
  const size_t szQ = (size_t)BN * 144 * n1;       // 18.87M
  const size_t szP = (size_t)BN * 254 * n1;       // 33.29M
  const size_t szC2 = (size_t)BN * CCH * n2;      // 1.57M

  float* ws = (float*)d_ws;
  float* t0   = ws;            ws += szC;   // ln out  (GEMM X input: over-read pads into t1)
  float* t1   = ws;            ws += szQ;   // qkv conv / geglu output (GEMM X input)
  float* t2   = ws;            ws += szQ;   // qkv dw
  float* t3   = ws;            ws += szC;   // attn out / gfm comb (GEMM X input)
  float* xa   = ws;            ws += szC;   // x + attn residual
  float* t4   = ws;            ws += szP;   // pin out
  float* t5   = ws;            ws += szP;   // ffn dw out
  float* b1   = ws;            ws += szC;   // branch1
  float* Abuf = ws;            ws += szC2;
  float* Bbuf = ws;            ws += szC2;
  float* Cbuf = ws;            ws += szC2;
  float* Dbuf = ws;            ws += szC2;
  float* b2a  = ws;            ws += szC2;  // tblock(A)
  float* b2   = ws;            ws += szC;   // branch2 (full res)
  float* nrm  = ws;            ws += BN * 96;
  float* sdot = ws;            ws += BN * HEADS * 36;
  float* attn = ws;            ws += BN * HEADS * 36;
  float* avgb = ws;            ws += BN * 96;
  float* mxb  = ws;            ws += BN * 96;
  float* cwb  = ws;            ws += BN * 96;
  float* savg = ws;            ws += (size_t)BN * n1;
  float* smax = ws;            ws += (size_t)BN * n1;
  float* samap= ws;            ws += (size_t)BN * n1;

  auto tblock = [&](const float* xin, float* outb, int n, int Hh, int Ww) {
    k_layernorm<<<dim3(n / 256, BN), 256, 0, stream>>>(xin, ln1_w, ln1_b, t0, n);
    k_gemm_wmma<<<dim3(n / 256, 9, BN), 256, 0, stream>>>(qkv_w, t0, nullptr, t1, 144, 48, n);
    k_dwconv3<<<dim3(n / 256, 144, BN), 256, 0, stream>>>(t1, qkv_dw_w, t2, 144, Hh, Ww);
    k_rownorm<<<BN * 96, 256, 0, stream>>>(t2, nrm, n);
    k_qkdot<<<dim3(36, HEADS, BN), 256, 0, stream>>>(t2, sdot, n);
    k_softmax<<<dim3(HEADS, BN), 32, 0, stream>>>(sdot, nrm, temp, attn);
    k_attnv<<<dim3(n / 256, CCH, BN), 256, 0, stream>>>(t2, attn, t3, n);
    k_gemm_wmma<<<dim3(n / 256, 3, BN), 256, 0, stream>>>(proj_w, t3, xin, xa, 48, 48, n);
    k_layernorm<<<dim3(n / 256, BN), 256, 0, stream>>>(xa, ln2_w, ln2_b, t0, n);
    k_gemm_wmma<<<dim3(n / 256, 16, BN), 256, 0, stream>>>(pin_w, t0, nullptr, t4, 254, 48, n);
    k_dwconv3<<<dim3(n / 256, 254, BN), 256, 0, stream>>>(t4, ffn_dw_w, t5, 254, Hh, Ww);
    k_geglu<<<dim3(n / 256, HID, BN), 256, 0, stream>>>(t5, t1, n);
    k_gemm_wmma<<<dim3(n / 256, 3, BN), 256, 0, stream>>>(pout_w, t1, xa, outb, 48, HID, n);
  };

  // branch 1: transformer block at full resolution
  tblock(x, b1, n1, 256, 256);

  // branch 2: wavelet -> transformer block on LL band -> inverse wavelet
  k_nlwt<<<dim3(n2 / 256, CCH, BN), 256, 0, stream>>>(x, Abuf, Bbuf, Cbuf, Dbuf, h1, h1);
  tblock(Abuf, b2a, n2, 128, 128);
  k_inlwt<<<dim3(n2 / 256, CCH, BN), 256, 0, stream>>>(b2a, Bbuf, Cbuf, Dbuf, b2, h1, h1);

  // GFM fusion
  k_chanreduce<<<BN * 96, 256, 0, stream>>>(b1, b2, avgb, mxb, n1);
  k_gfm_mlp<<<BN, 96, 0, stream>>>(avgb, mxb, ca_w1, ca_w2, cwb);
  k_gfm_combine<<<dim3(n1 / 256, BN), 256, 0, stream>>>(b1, b2, cwb, t3, savg, smax, n1);
  k_saconv<<<dim3(n1 / 256, BN), 256, 0, stream>>>(savg, smax, sa_w, sa_b, samap, 256, 256);
  k_final<<<dim3(n1 / 256, CCH, BN), 256, 0, stream>>>(t3, samap, outp, n1);
}
